// GNNModel_21320217657455
// MI455X (gfx1250) — matrix operand order repaired and checked by reading, not compiled, not run on hardware
//
#include <hip/hip_runtime.h>
#include <hip/hip_bf16.h>

typedef __attribute__((ext_vector_type(16))) __bf16 v16bf;
typedef __attribute__((ext_vector_type(8)))  float  v8f;

#define WAVE 32

// ---------------------------------------------------------------------------
// Degree / normalization kernels
// ---------------------------------------------------------------------------
__global__ void deg_init_kernel(float* __restrict__ deg, int n) {
    int i = blockIdx.x * blockDim.x + threadIdx.x;
    if (i < n) deg[i] = 1.0f;  // self-loop contributes 1
}

__global__ void deg_edge_kernel(const int* __restrict__ dst, float* __restrict__ deg, int E) {
    int e = blockIdx.x * blockDim.x + threadIdx.x;
    if (e < E) atomicAdd(&deg[dst[e]], 1.0f);
}

__global__ void deg_rsqrt_kernel(float* __restrict__ deg, int n) {
    int i = blockIdx.x * blockDim.x + threadIdx.x;
    if (i < n) deg[i] = rsqrtf(deg[i]);  // deg >= 1 always (self loop)
}

// ---------------------------------------------------------------------------
// Register-blocked bf16 WMMA GEMM with LDS-staged, pre-swizzled weights.
//
//   C[M, Nn] = A[M, K] * B[K, Nn]     (A, B f32 in memory; bf16 in-register,
//                                       f32 accumulate)
//
// Each block: 8 waves; each wave owns a 16-row strip and computes NB (=8)
// 16x16 output tiles across a 128-wide column block (blockIdx.y selects the
// column block). The weight slice (K x 128) is converted to bf16 and stored
// in LDS pre-swizzled into the WMMA B operand layout, so each lane fetches
// its 16 bf16 B values as one contiguous 32B chunk (2x ds_load_b128).
//
// Per K-step, ALL NB B-tiles are loaded into a register array first (single
// ds clause + single s_wait_dscnt), then the NB WMMAs issue back-to-back on
// the matrix pipe while the next A global_load clause is in flight.
// ---------------------------------------------------------------------------
template <int K, int NB>
__launch_bounds__(256)
__global__ void gemm_bf16_wmma_kernel(const float* __restrict__ A,
                                      const float* __restrict__ B,
                                      float* __restrict__ C,
                                      int M, int Nn) {
    extern __shared__ __align__(16) unsigned char smem[];  // (K/32)*NB*1024 bytes

    const int tid     = threadIdx.x;
    const int lane    = tid & (WAVE - 1);
    const int wv      = tid >> 5;
    const int colBase = blockIdx.y * (NB * 16);

    // ---- Stage weight slice into LDS, swizzled to WMMA B layout ----
    const int totalElems = K * NB * 16;
    for (int idx = tid; idx < totalElems; idx += 256) {
        int k = idx / (NB * 16);
        int n = idx % (NB * 16);
        float v = B[(size_t)k * Nn + colBase + n];
        int kt = k >> 5, kk = k & 31;
        int nt = n >> 4, nn = n & 15;
        int ln = (((kk >> 3) & 1) ? 16 : 0) + nn;
        int j  = (kk & 7) | ((kk >> 4) << 3);
        *(__bf16*)(smem + ((size_t)(kt * NB + nt) * 1024) + ln * 32 + j * 2) = (__bf16)v;
    }
    __syncthreads();

    const int mt = blockIdx.x * 8 + wv;          // 16-row strip index
    if (mt * 16 >= M) return;                    // wave-uniform: EXEC stays all-ones

    // A operand layout: lane<16 -> row lane&15, K = {0..7,16..23}+k0;
    //                   lane>=16 -> K = {8..15,24..31}+k0.
    const int aRow  = mt * 16 + (lane & 15);
    const int aKoff = (lane >= 16) ? 8 : 0;
    const float* ap0 = A + (size_t)aRow * K + aKoff;

    v8f acc[NB];
#pragma unroll
    for (int t = 0; t < NB; ++t) acc[t] = (v8f){};

#pragma unroll
    for (int kt = 0; kt < K / 32; ++kt) {
        const float* ap = ap0 + kt * 32;
        v16bf a;
#pragma unroll
        for (int j = 0; j < 8; ++j) {
            a[j]     = (__bf16)ap[j];        // K = kt*32 + aKoff + j
            a[j + 8] = (__bf16)ap[16 + j];   // K = kt*32 + aKoff + 16 + j
        }
        // Batch-load all NB B tiles (one ds clause, one wait)
        v16bf b[NB];
#pragma unroll
        for (int t = 0; t < NB; ++t) {
            b[t] = *(const v16bf*)(smem + ((size_t)(kt * NB + t) * 1024) + lane * 32);
        }
        // Back-to-back WMMAs on the matrix pipe
#pragma unroll
        for (int t = 0; t < NB; ++t) {
            acc[t] = __builtin_amdgcn_wmma_f32_16x16x32_bf16(
                /*neg_a=*/false, a, /*neg_b=*/false, b[t],
                /*c_mod=*/(short)0, acc[t], /*reuse_a=*/false, /*reuse_b=*/false);
        }
    }

    // C/D layout: VGPR r -> lanes 0-15: (M=r, N=lane); lanes 16-31: (M=r+8, N=lane-16)
    const int cRow     = mt * 16 + ((lane >= 16) ? 8 : 0);
    const int cColBase = colBase + (lane & 15);
#pragma unroll
    for (int t = 0; t < NB; ++t) {
#pragma unroll
        for (int r = 0; r < 8; ++r) {
            C[(size_t)(cRow + r) * Nn + cColBase + t * 16] = acc[t][r];
        }
    }
}

// ---------------------------------------------------------------------------
// agg[i,f] = h[i,f] * dinv[i]^2 + bias[f]   (folds self-loop message + bias)
// ---------------------------------------------------------------------------
__global__ void agg_init_kernel(const float* __restrict__ h,
                                const float* __restrict__ dinv,
                                const float* __restrict__ bias,
                                float* __restrict__ agg,
                                int n, int F) {
    long long idx = (long long)blockIdx.x * blockDim.x + threadIdx.x;
    long long tot = (long long)n * F;
    if (idx >= tot) return;
    int i = (int)(idx / F);
    int f = (int)(idx % F);
    float w = dinv[i] * dinv[i];
    agg[idx] = h[idx] * w + bias[f];
}

// ---------------------------------------------------------------------------
// Edge scatter-add: one wave per edge; lanes cover F features with float4.
// agg[dst] += h[src] * (dinv[src]*dinv[dst])
// ---------------------------------------------------------------------------
template <int F>
__launch_bounds__(256)
__global__ void edge_scatter_kernel(const int* __restrict__ src,
                                    const int* __restrict__ dst,
                                    const float* __restrict__ dinv,
                                    const float* __restrict__ h,
                                    float* __restrict__ agg,
                                    int E) {
    const int lane = threadIdx.x & (WAVE - 1);
    const int wid  = (blockIdx.x * (blockDim.x / WAVE)) + (threadIdx.x / WAVE);
    if (wid >= E) return;
    const int s = src[wid];
    const int d = dst[wid];
    const float w = dinv[s] * dinv[d];
    const float4* hp = (const float4*)(h + (size_t)s * F);
    float* ap = agg + (size_t)d * F;
#pragma unroll
    for (int c = 0; c < F / 128; ++c) {
        float4 v = hp[c * 32 + lane];
        int base = c * 128 + lane * 4;
        atomicAdd(ap + base + 0, v.x * w);
        atomicAdd(ap + base + 1, v.y * w);
        atomicAdd(ap + base + 2, v.z * w);
        atomicAdd(ap + base + 3, v.w * w);
    }
}

__global__ void relu_kernel(float* __restrict__ a, long long n) {
    long long idx = (long long)blockIdx.x * blockDim.x + threadIdx.x;
    if (idx < n) a[idx] = fmaxf(a[idx], 0.0f);
}

// ---------------------------------------------------------------------------
// Launch
// ---------------------------------------------------------------------------
extern "C" void kernel_launch(void* const* d_in, const int* in_sizes, int n_in,
                              void* d_out, int out_size, void* d_ws, size_t ws_size,
                              hipStream_t stream) {
    (void)n_in; (void)out_size; (void)ws_size;

    const int NUM_CLASSES = 256;
    const int HIDDEN      = 128;

    const float* x  = (const float*)d_in[0];
    const int*   ei = (const int*)d_in[1];
    const float* W1 = (const float*)d_in[2];
    const float* b1 = (const float*)d_in[3];
    const float* W2 = (const float*)d_in[4];
    const float* b2 = (const float*)d_in[5];

    const int N = in_sizes[0] / NUM_CLASSES;  // 100000
    const int E = in_sizes[1] / 2;            // 1600000
    const int* src = ei;
    const int* dst = ei + E;

    float* out = (float*)d_out;

    // Workspace layout (floats)
    float* ws    = (float*)d_ws;
    size_t nAl   = ((size_t)N + 255) & ~(size_t)255;
    float* dinv  = ws;                               // N
    float* h0    = dinv + nAl;                       // N*128  (GEMM1 out)
    float* agg1  = h0 + (size_t)N * HIDDEN;          // N*128  (agg -> h1)
    float* h2    = agg1 + (size_t)N * HIDDEN;        // N*256  (GEMM2 out)

    const int TB = 256;
    const int wavesPerBlock = TB / WAVE;

    // 1) degrees (with self loop), dinv = rsqrt(deg)
    deg_init_kernel<<<(N + TB - 1) / TB, TB, 0, stream>>>(dinv, N);
    deg_edge_kernel<<<(E + TB - 1) / TB, TB, 0, stream>>>(dst, dinv, E);
    deg_rsqrt_kernel<<<(N + TB - 1) / TB, TB, 0, stream>>>(dinv, N);

    // 2) h0 = x @ W1   (100000x256 @ 256x128)
    {
        int mTiles   = N / 16;                          // 6250
        int blocksX  = (mTiles + 7) / 8;                // 8 waves/block
        dim3 grid(blocksX, HIDDEN / 128, 1);            // y = 1 column block
        size_t smem = (size_t)(256 / 32) * 8 * 1024;    // 64 KB
        gemm_bf16_wmma_kernel<256, 8><<<grid, TB, smem, stream>>>(x, W1, h0, N, HIDDEN);
    }

    // 3) agg1 = h0 * dinv^2 + b1 (self loop + bias), then edge scatter-add
    {
        long long tot = (long long)N * HIDDEN;
        agg_init_kernel<<<(int)((tot + TB - 1) / TB), TB, 0, stream>>>(h0, dinv, b1, agg1, N, HIDDEN);
        int blocks = (E + wavesPerBlock - 1) / wavesPerBlock;
        edge_scatter_kernel<HIDDEN><<<blocks, TB, 0, stream>>>(src, dst, dinv, h0, agg1, E);
        relu_kernel<<<(int)((tot + TB - 1) / TB), TB, 0, stream>>>(agg1, tot);
    }

    // 4) h2 = h1 @ W2   (100000x128 @ 128x256)
    {
        int mTiles   = N / 16;
        int blocksX  = (mTiles + 7) / 8;
        dim3 grid(blocksX, NUM_CLASSES / 128, 1);       // y = 2 column blocks
        size_t smem = (size_t)(128 / 32) * 8 * 1024;    // 32 KB
        gemm_bf16_wmma_kernel<128, 8><<<grid, TB, smem, stream>>>(agg1, W2, h2, N, NUM_CLASSES);
    }

    // 5) out = h2 * dinv^2 + b2, then edge scatter-add into out
    {
        long long tot = (long long)N * NUM_CLASSES;
        agg_init_kernel<<<(int)((tot + TB - 1) / TB), TB, 0, stream>>>(h2, dinv, b2, out, N, NUM_CLASSES);
        int blocks = (E + wavesPerBlock - 1) / wavesPerBlock;
        edge_scatter_kernel<NUM_CLASSES><<<blocks, TB, 0, stream>>>(src, dst, dinv, h2, out, E);
    }
}
